// DMOT_62775241998644
// MI455X (gfx1250) — matrix operand order repaired and checked by reading, not compiled, not run on hardware
//
#include <hip/hip_runtime.h>
#include <math.h>
#include <stdint.h>

typedef __attribute__((ext_vector_type(16))) _Float16 v16h;
typedef __attribute__((ext_vector_type(8)))  float    v8f;

// ---------------- problem constants ----------------
#define BB    16
#define CIN   384
#define HW0   4096          // 64*64
#define CDIMC 128
#define KK    64
#define M1    1152          // 512 (feat) + 512 (score) + 128 (dustbin) fused first layer

// ---------------- workspace layout (bytes, all 256-aligned) ----------------
static constexpr size_t OFF_XT  = 0;           // [4096,384] f16     (3,145,728)
static constexpr size_t OFF_W1  = 3145728;     // 72x12 A-frags f16  (  884,736)
static constexpr size_t OFF_W2L = 4030464;     // 8x16 A-frags       (  131,072)
static constexpr size_t OFF_W2S = 4161536;     // 4x16 A-frags       (   65,536)
static constexpr size_t OFF_W2D = 4227072;     // 1x4  A-frags       (    4,096)
static constexpr size_t OFF_B1  = 4231168;     // 1152 f32           (    4,608)
static constexpr size_t OFF_BD  = 4235776;     // 16 f32 (padded)    (      256)
static constexpr size_t OFF_HT  = 4236032;     // [4096,1152] f16    (9,437,184)
static constexpr size_t OFF_LF  = 13673216;    // [128,4096] f16     (1,048,576)
static constexpr size_t OFF_Z   = 14721792;    // [65,4096] f32      (1,064,960)
static constexpr size_t OFF_P   = 15786752;    // [64,4096] f16      (  524,288)
static constexpr size_t OFF_OUT = 16311040;    // [128,64] f32       (   32,768)
static constexpr size_t OFF_ACC = 16343808;    // [16,8192] f32      (  524,288)
// total ~16.9 MB

// ---------------- utility kernels ----------------
__global__ void k_zero(float* __restrict__ p, int n) {
  int i = blockIdx.x * blockDim.x + threadIdx.x;
  if (i < n) p[i] = 0.f;
}

// Swizzle a row-major [M_real, K] f32 weight into WMMA A-fragment order:
// fragment (mt,kt) = 512 contiguous f16 [lane*16 + i]; lane-half selects K run.
__global__ void k_swizzle_a(const float* __restrict__ W, int M_real,
                            int Mtiles, int Ktiles, int K,
                            _Float16* __restrict__ out) {
  int t = blockIdx.x * blockDim.x + threadIdx.x;
  int total = Mtiles * Ktiles * 512;
  if (t >= total) return;
  int frag = t >> 9;
  int r    = t & 511;
  int lane = r >> 4, i = r & 15;
  int mt = frag / Ktiles, kt = frag % Ktiles;
  int half_id = lane >> 4;
  int m = mt * 16 + (lane & 15);
  int k = kt * 32 + half_id * 16 + i;
  float v = (m < M_real) ? W[(size_t)m * K + k] : 0.f;
  out[t] = (_Float16)v;
}

__global__ void k_bias_cat(const float* __restrict__ a, const float* __restrict__ b,
                           const float* __restrict__ c, float* __restrict__ dst) {
  int i = blockIdx.x * blockDim.x + threadIdx.x;
  if (i >= M1) return;
  dst[i] = (i < 512) ? a[i] : ((i < 1024) ? b[i - 512] : c[i - 1024]);
}

__global__ void k_bias_pad16(const float* __restrict__ s, float* __restrict__ dst) {
  int i = threadIdx.x;
  if (i < 16) dst[i] = (i == 0) ? s[0] : 0.f;
}

// x[b]: [384, 64, 64] f32 -> xt: [Np, 384] f16 (optionally 2x2 avg-pooled)
__global__ void k_xt(const float* __restrict__ x, _Float16* __restrict__ xt,
                     int pool, int Np) {
  int idx = blockIdx.x * blockDim.x + threadIdx.x;
  if (idx >= CIN * Np) return;
  int c = idx / Np, p = idx - c * Np;
  const float* xc = x + (size_t)c * HW0;
  float v;
  if (!pool) {
    v = xc[p];
  } else {
    int h = p >> 5, w = p & 31;
    int b0 = (h << 7) + (w << 1);            // (2h)*64 + 2w
    v = 0.25f * (xc[b0] + xc[b0 + 1] + xc[b0 + 64] + xc[b0 + 65]);
  }
  xt[(size_t)p * CIN + c] = (_Float16)v;
}

// ---------------- generic WMMA GEMM, MB-way M register blocking ------------
// D[M, Ncols] = A[M,K] * B[K, Ncols] + bias;   A pre-swizzled fragments,
// B stored activation-major: Bm[n*ldb + koff + k] f16 (K contiguous).
// One wave computes MB vertically-adjacent 16x16 tiles, re-using each B
// fragment for MB WMMAs (4x less activation traffic).
// mode 0: relu -> f16, transposed out[n*ldo + m]     (hidden activations)
// mode 1: f16 row-major out[m*ldo + n]               (local features)
// mode 2: f32 row-major out[m*ldo + n], rows<M_real  (scores / dustbin)
template <int MB>
__global__ void k_wmma_gemm(const _Float16* __restrict__ Af,
                            const _Float16* __restrict__ Bm, int ldb, int koff,
                            const float* __restrict__ bias,
                            void* __restrict__ outp,
                            int Mtiles, int Ntn, int Ktiles,
                            int mode, int M_real, int ldo) {
  int wid = (blockIdx.x * blockDim.x + threadIdx.x) >> 5;
  const int Mg = Mtiles / MB;
  if (wid >= Mg * Ntn) return;                  // wave-uniform: EXEC stays full
  const int lane    = threadIdx.x & 31;
  const int half_id = lane >> 4;
  const int n_in    = lane & 15;
  const int mg = wid / Ntn, nt = wid % Ntn;

  v8f acc[MB];
#pragma unroll
  for (int j = 0; j < MB; ++j) acc[j] = (v8f){};

  const size_t astride = (size_t)Ktiles * 512;
  const _Float16* ap = Af + (size_t)(mg * MB) * astride + lane * 16;
  const _Float16* bp = Bm + (size_t)(nt * 16 + n_in) * ldb + koff + half_id * 16;
  for (int kt = 0; kt < Ktiles; ++kt) {
    v16h b = *(const v16h*)bp;
    bp += 32;
    __builtin_prefetch(bp, 0, 3);
#pragma unroll
    for (int j = 0; j < MB; ++j) {
      v16h a = *(const v16h*)(ap + j * astride);
      acc[j] = __builtin_amdgcn_wmma_f32_16x16x32_f16(false, a, false, b,
                                                      (short)0, acc[j], false, false);
    }
    ap += 512;
  }

  const int col = nt * 16 + n_in;
#pragma unroll
  for (int j = 0; j < MB; ++j) {
    const int row0 = (mg * MB + j) * 16 + half_id * 8;
    if (mode == 0) {
      union { uint4 q; _Float16 h[8]; } u;
#pragma unroll
      for (int v = 0; v < 8; ++v) {
        float t = acc[j][v] + bias[row0 + v];
        u.h[v] = (_Float16)fmaxf(t, 0.f);
      }
      *(uint4*)((_Float16*)outp + (size_t)col * ldo + row0) = u.q;
    } else if (mode == 1) {
      _Float16* o = (_Float16*)outp;
#pragma unroll
      for (int v = 0; v < 8; ++v)
        o[(size_t)(row0 + v) * ldo + col] = (_Float16)(acc[j][v] + bias[row0 + v]);
    } else {
      float* o = (float*)outp;
#pragma unroll
      for (int v = 0; v < 8; ++v) {
        int r = row0 + v;
        if (r < M_real) o[(size_t)r * ldo + col] = acc[j][v] + bias[r];
      }
    }
  }
}

// ---------------- Sinkhorn (log optimal transport), one block per image ----
__global__ void __launch_bounds__(1024)
k_sinkhorn(const float* __restrict__ Z, _Float16* __restrict__ P, int N) {
  __shared__ float v_sh[4096];
  __shared__ float u_sh[72];
  __shared__ float rs_sh[64];
  const int tid   = threadIdx.x;
  const int lane  = tid & 31;
  const int warp  = tid >> 5;
  const int nwarp = blockDim.x >> 5;
  const float log_mu = -logf((float)(KK + 1));
  const float log_nu = -logf((float)N);

  for (int n = tid; n < N; n += blockDim.x) v_sh[n] = 0.f;
  __syncthreads();

  for (int it = 0; it < 3; ++it) {
    // u[i] = log_mu - LSE_n(Z[i,n] + v[n]) : one wave32 per row
    for (int i = warp; i < KK + 1; i += nwarp) {
      const float* zr = Z + (size_t)i * N;
      float m = -INFINITY, s = 0.f;
      for (int n = lane; n < N; n += 32) {
        float z = zr[n] + v_sh[n];
        if (z > m) { s = s * expf(m - z) + 1.f; m = z; }
        else       { s += expf(z - m); }
      }
      for (int off = 16; off; off >>= 1) {
        float mo = __shfl_xor(m, off, 32);
        float so = __shfl_xor(s, off, 32);
        float mn = fmaxf(m, mo);
        s = s * expf(m - mn) + so * expf(mo - mn);
        m = mn;
      }
      if (lane == 0) u_sh[i] = log_mu - (m + logf(s));
    }
    __syncthreads();
    // v[n] = log_nu - LSE_i(Z[i,n] + u[i])
    for (int n = tid; n < N; n += blockDim.x) {
      float m = -INFINITY, s = 0.f;
      for (int i = 0; i < KK + 1; ++i) {
        float z = Z[(size_t)i * N + n] + u_sh[i];
        if (z > m) { s = s * expf(m - z) + 1.f; m = z; }
        else       { s += expf(z - m); }
      }
      v_sh[n] = log_nu - (m + logf(s));
    }
    __syncthreads();
  }

  // row sums of exp(Z+u+v) over first 64 rows
  for (int k = warp; k < KK; k += nwarp) {
    const float* zr = Z + (size_t)k * N;
    float uk = u_sh[k];
    float s = 0.f;
    for (int n = lane; n < N; n += 32) s += expf(zr[n] + uk + v_sh[n]);
    for (int off = 16; off; off >>= 1) s += __shfl_xor(s, off, 32);
    if (lane == 0) rs_sh[k] = s;
  }
  __syncthreads();
  for (int idx = tid; idx < KK * N; idx += blockDim.x) {
    int k = idx / N, n = idx - k * N;
    float p = expf(Z[(size_t)k * N + n] + u_sh[k] + v_sh[n]) / (rs_sh[k] + 1e-8f);
    P[idx] = (_Float16)p;
  }
}

// ---------------- descriptor einsum: out[c,k] = sum_n lf[c,n] * P[k,n] -----
__global__ void k_wmma_einsum(const _Float16* __restrict__ lf,
                              const _Float16* __restrict__ P,
                              int N, float* __restrict__ outm) {
  int wid = (blockIdx.x * blockDim.x + threadIdx.x) >> 5;
  if (wid >= (CDIMC / 16) * (KK / 16)) return;
  const int lane    = threadIdx.x & 31;
  const int half_id = lane >> 4;
  const int n_in    = lane & 15;
  const int mt = wid >> 2, nt = wid & 3;
  const int Kt = N >> 5;

  v8f acc = {};
  const _Float16* ap = lf + (size_t)(mt * 16 + n_in) * N + half_id * 16;
  const _Float16* bp = P  + (size_t)(nt * 16 + n_in) * N + half_id * 16;
  for (int kt = 0; kt < Kt; ++kt) {
    union { v16h v; uint4 q[2]; } ua, ub;
    ua.q[0] = *(const uint4*)(ap);         // K = kt*32 + half*16 .. +7
    ua.q[1] = *(const uint4*)(ap + 8);     // .. +15
    ub.q[0] = *(const uint4*)(bp);         // K = kt*32 + half*16 .. +7
    ub.q[1] = *(const uint4*)(bp + 8);     // .. +15
    ap += 32;
    bp += 32;
    acc = __builtin_amdgcn_wmma_f32_16x16x32_f16(false, ua.v, false, ub.v,
                                                 (short)0, acc, false, false);
  }
  const int row0 = mt * 16 + half_id * 8;
#pragma unroll
  for (int v = 0; v < 8; ++v)
    outm[(row0 + v) * KK + nt * 16 + n_in] = acc[v];
}

// column L2-normalize out[128,64] and accumulate into per-image descriptor
__global__ void k_accnorm(const float* __restrict__ outm, float* __restrict__ acc) {
  int k = threadIdx.x;
  if (k >= KK) return;
  float ss = 0.f;
  for (int c = 0; c < CDIMC; ++c) { float v = outm[c * KK + k]; ss += v * v; }
  float nrm = fmaxf(sqrtf(ss), 1e-12f);
  for (int c = 0; c < CDIMC; ++c) acc[c * KK + k] += outm[c * KK + k] / nrm;
}

// g = mean over scales, L2-normalized per image
__global__ void k_final(const float* __restrict__ acc, float* __restrict__ g) {
  int b = blockIdx.x;
  const float* a = acc + (size_t)b * (CDIMC * KK);
  __shared__ float red[256];
  float ss = 0.f;
  for (int j = threadIdx.x; j < CDIMC * KK; j += 256) {
    float v = a[j] * 0.5f;
    ss += v * v;
  }
  red[threadIdx.x] = ss;
  __syncthreads();
  for (int o = 128; o; o >>= 1) {
    if (threadIdx.x < o) red[threadIdx.x] += red[threadIdx.x + o];
    __syncthreads();
  }
  float nrm = fmaxf(sqrtf(red[0]), 1e-12f);
  for (int j = threadIdx.x; j < CDIMC * KK; j += 256)
    g[(size_t)b * (CDIMC * KK) + j] = (a[j] * 0.5f) / nrm;
}

// ---------------- host orchestration ----------------
extern "C" void kernel_launch(void* const* d_in, const int* in_sizes, int n_in,
                              void* d_out, int out_size, void* d_ws, size_t ws_size,
                              hipStream_t stream) {
  (void)in_sizes; (void)n_in; (void)out_size; (void)ws_size;  // needs ~17 MB ws
  const float* x = (const float*)d_in[0];
  uint8_t* ws = (uint8_t*)d_ws;
  _Float16* w_xt  = (_Float16*)(ws + OFF_XT);
  _Float16* w_w1  = (_Float16*)(ws + OFF_W1);
  _Float16* w_w2l = (_Float16*)(ws + OFF_W2L);
  _Float16* w_w2s = (_Float16*)(ws + OFF_W2S);
  _Float16* w_w2d = (_Float16*)(ws + OFF_W2D);
  float*    w_b1  = (float*)(ws + OFF_B1);
  float*    w_bd  = (float*)(ws + OFF_BD);
  _Float16* w_ht  = (_Float16*)(ws + OFF_HT);
  _Float16* w_lf  = (_Float16*)(ws + OFF_LF);
  float*    w_Z   = (float*)(ws + OFF_Z);
  _Float16* w_P   = (_Float16*)(ws + OFF_P);
  float*    w_out = (float*)(ws + OFF_OUT);
  float*    w_acc = (float*)(ws + OFF_ACC);

  k_zero<<<(BB * CDIMC * KK + 255) / 256, 256, 0, stream>>>(w_acc, BB * CDIMC * KK);

  for (int s = 0; s < 2; ++s) {
    const int base = 1 + 12 * s;
    const float* fw1 = (const float*)d_in[base + 0];
    const float* fb1 = (const float*)d_in[base + 1];
    const float* fw2 = (const float*)d_in[base + 2];
    const float* fb2 = (const float*)d_in[base + 3];
    const float* sw1 = (const float*)d_in[base + 4];
    const float* sb1 = (const float*)d_in[base + 5];
    const float* sw2 = (const float*)d_in[base + 6];
    const float* sb2 = (const float*)d_in[base + 7];
    const float* dw1 = (const float*)d_in[base + 8];
    const float* db1 = (const float*)d_in[base + 9];
    const float* dw2 = (const float*)d_in[base + 10];
    const float* db2 = (const float*)d_in[base + 11];

    auto sw = [&](const float* W, int Mreal, int Mt, int Ktl, int K, _Float16* o) {
      int tot = Mt * Ktl * 512;
      k_swizzle_a<<<(tot + 255) / 256, 256, 0, stream>>>(W, Mreal, Mt, Ktl, K, o);
    };
    sw(fw1, 512, 32, 12, 384, w_w1);
    sw(sw1, 512, 32, 12, 384, w_w1 + (size_t)32 * 12 * 512);
    sw(dw1, 128,  8, 12, 384, w_w1 + (size_t)64 * 12 * 512);
    sw(fw2, 128,  8, 16, 512, w_w2l);
    sw(sw2,  64,  4, 16, 512, w_w2s);
    sw(dw2,   1,  1,  4, 128, w_w2d);
    k_bias_cat<<<(M1 + 255) / 256, 256, 0, stream>>>(fb1, sb1, db1, w_b1);
    k_bias_pad16<<<1, 16, 0, stream>>>(db2, w_bd);

    const int N   = (s == 0) ? 4096 : 1024;
    const int Ntn = N / 16;
    for (int b = 0; b < BB; ++b) {
      const float* xb = x + (size_t)b * CIN * HW0;
      k_xt<<<(CIN * N + 255) / 256, 256, 0, stream>>>(xb, w_xt, s, N);

      auto gemm = [&](const _Float16* Af, const _Float16* Bm, int ldb, int koff,
                      const float* bias, void* o, int Mt, int Ktl,
                      int mode, int Mreal, int ldo) {
        if (Mt % 4 == 0) {
          int waves = (Mt / 4) * Ntn;
          k_wmma_gemm<4><<<(waves * 32 + 255) / 256, 256, 0, stream>>>(
              Af, Bm, ldb, koff, bias, o, Mt, Ntn, Ktl, mode, Mreal, ldo);
        } else {
          int waves = Mt * Ntn;
          k_wmma_gemm<1><<<(waves * 32 + 255) / 256, 256, 0, stream>>>(
              Af, Bm, ldb, koff, bias, o, Mt, Ntn, Ktl, mode, Mreal, ldo);
        }
      };
      // fused first layer: ht[p,1152] = relu(W1 * x + b1)
      gemm(w_w1, w_xt, CIN, 0, w_b1, w_ht, 72, 12, 0, M1, M1);
      // local features: lf[128,N]
      gemm(w_w2l, w_ht, M1, 0, fb2, w_lf, 8, 16, 1, CDIMC, N);
      // scores -> Z rows 0..63
      gemm(w_w2s, w_ht, M1, 512, sb2, w_Z, 4, 16, 2, KK, N);
      // dustbin -> Z row 64
      gemm(w_w2d, w_ht, M1, 1024, w_bd, w_Z + (size_t)KK * N, 1, 4, 2, 1, N);

      k_sinkhorn<<<1, 1024, 0, stream>>>(w_Z, w_P, N);
      k_wmma_einsum<<<4, 256, 0, stream>>>(w_lf, w_P, N, w_out);
      k_accnorm<<<1, 64, 0, stream>>>(w_out, w_acc + (size_t)b * (CDIMC * KK));
    }
  }
  k_final<<<BB, 256, 0, stream>>>(w_acc, (float*)d_out);
}
